// BidirectionalFlowEstimation_4466765988256
// MI455X (gfx1250) — compile-verified, matrix operand order checked by reading, not run on hardware
//
#include <hip/hip_runtime.h>

typedef __attribute__((ext_vector_type(16))) _Float16 v16h;
typedef __attribute__((ext_vector_type(8)))  _Float16 v8h;
typedef __attribute__((ext_vector_type(8)))  float    v8f;
typedef __attribute__((ext_vector_type(4)))  float    v4f;

constexpr int  BN   = 8;
constexpr int  H    = 256;
constexpr int  W    = 256;
constexpr int  C    = 32;
constexpr long NPIX = (long)BN * H * W;      // 524288
constexpr int  HO   = 512;
constexpr int  WO   = 512;

// ---------------------------------------------------------------------------
// Warp (bilinear, TFA semantics) + concat(other, warped) -> f16 NHWC[64]
// one thread per (pixel, group of 8 channels)
// ---------------------------------------------------------------------------
__global__ void __launch_bounds__(256) warp_concat_kernel(
    const float* __restrict__ src, const float* __restrict__ oth,
    const float* __restrict__ flw, _Float16* __restrict__ act0) {
  long idx = (long)blockIdx.x * blockDim.x + threadIdx.x;
  long p = idx >> 2;
  int  g = (int)(idx & 3);
  if (p >= NPIX) return;
  int wx = (int)(p % W);
  int hy = (int)((p / W) % H);
  int bb = (int)(p / ((long)W * H));
  float qy = (float)hy - flw[p * 2 + 0];
  float qx = (float)wx - flw[p * 2 + 1];
  float fy = fminf(fmaxf(floorf(qy), 0.f), (float)(H - 2));
  float fx = fminf(fmaxf(floorf(qx), 0.f), (float)(W - 2));
  float ay = fminf(fmaxf(qy - fy, 0.f), 1.f);
  float ax = fminf(fmaxf(qx - fx, 0.f), 1.f);
  int iy = (int)fy, ix = (int)fx;
  const float* b0 = src + (((long)bb * H + iy) * W + ix) * C + g * 8;
  const float* b1 = b0 + (long)W * C;
  v4f tl0 = *(const v4f*)(b0);       v4f tl1 = *(const v4f*)(b0 + 4);
  v4f tr0 = *(const v4f*)(b0 + C);   v4f tr1 = *(const v4f*)(b0 + C + 4);
  v4f bl0 = *(const v4f*)(b1);       v4f bl1 = *(const v4f*)(b1 + 4);
  v4f br0 = *(const v4f*)(b1 + C);   v4f br1 = *(const v4f*)(b1 + C + 4);
  v4f top0 = tl0 + ax * (tr0 - tl0), top1 = tl1 + ax * (tr1 - tl1);
  v4f bot0 = bl0 + ax * (br0 - bl0), bot1 = bl1 + ax * (br1 - bl1);
  v4f r0 = top0 + ay * (bot0 - top0), r1 = top1 + ay * (bot1 - top1);
  const float* op = oth + p * C + g * 8;
  v4f o0 = *(const v4f*)(op), o1 = *(const v4f*)(op + 4);
  v8h ho, hw;
#pragma unroll
  for (int j = 0; j < 4; ++j) {
    ho[j] = (_Float16)o0[j];  ho[j + 4] = (_Float16)o1[j];
    hw[j] = (_Float16)r0[j];  hw[j + 4] = (_Float16)r1[j];
  }
  *(v8h*)(act0 + p * 64 + g * 8)      = ho;   // channels [0..31]  = other image
  *(v8h*)(act0 + p * 64 + 32 + g * 8) = hw;   // channels [32..63] = warped image
}

// ---------------------------------------------------------------------------
// Second warp with refined flow -> f32 output
// ---------------------------------------------------------------------------
__global__ void __launch_bounds__(256) warp_out_kernel(
    const float* __restrict__ src, const float* __restrict__ flw,
    float* __restrict__ out) {
  long idx = (long)blockIdx.x * blockDim.x + threadIdx.x;
  long p = idx >> 2;
  int  g = (int)(idx & 3);
  if (p >= NPIX) return;
  int wx = (int)(p % W);
  int hy = (int)((p / W) % H);
  int bb = (int)(p / ((long)W * H));
  float qy = (float)hy - flw[p * 2 + 0];
  float qx = (float)wx - flw[p * 2 + 1];
  float fy = fminf(fmaxf(floorf(qy), 0.f), (float)(H - 2));
  float fx = fminf(fmaxf(floorf(qx), 0.f), (float)(W - 2));
  float ay = fminf(fmaxf(qy - fy, 0.f), 1.f);
  float ax = fminf(fmaxf(qx - fx, 0.f), 1.f);
  int iy = (int)fy, ix = (int)fx;
  const float* b0 = src + (((long)bb * H + iy) * W + ix) * C + g * 8;
  const float* b1 = b0 + (long)W * C;
  v4f tl0 = *(const v4f*)(b0);       v4f tl1 = *(const v4f*)(b0 + 4);
  v4f tr0 = *(const v4f*)(b0 + C);   v4f tr1 = *(const v4f*)(b0 + C + 4);
  v4f bl0 = *(const v4f*)(b1);       v4f bl1 = *(const v4f*)(b1 + 4);
  v4f br0 = *(const v4f*)(b1 + C);   v4f br1 = *(const v4f*)(b1 + C + 4);
  v4f top0 = tl0 + ax * (tr0 - tl0), top1 = tl1 + ax * (tr1 - tl1);
  v4f bot0 = bl0 + ax * (br0 - bl0), bot1 = bl1 + ax * (br1 - bl1);
  v4f r0 = top0 + ay * (bot0 - top0), r1 = top1 + ay * (bot1 - top1);
  *(v4f*)(out + p * C + g * 8)     = r0;
  *(v4f*)(out + p * C + g * 8 + 4) = r1;
}

// ---------------------------------------------------------------------------
// Pre-swizzle f32 HWIO weights into f16 B-fragment order:
// dst[(((chunk*(Cout/16)) + ntile)*32 + lane)*16 + e], so the conv kernel's
// B fragment is a single contiguous 32B LDS vector load per lane per K-chunk.
// ---------------------------------------------------------------------------
__global__ void __launch_bounds__(256) weight_swizzle_kernel(
    const float* __restrict__ w, _Float16* __restrict__ dst, int K, int Cout) {
  int idx = blockIdx.x * blockDim.x + threadIdx.x;
  if (idx >= K * Cout) return;
  int k = idx / Cout, n = idx % Cout;
  int chunk = k >> 5, kl = k & 31;
  int g  = kl >> 4;            // which 16-wide K half (VGPR 0-3 vs 4-7)
  int hs = (kl >> 3) & 1;      // lane group (0-15 vs 16-31)
  int r  = kl & 7;
  int e  = g * 8 + r;
  int lane = hs * 16 + (n & 15);
  int nt   = n >> 4;
  dst[(((long)chunk * (Cout >> 4) + nt) * 32 + lane) * 16 + e] = (_Float16)w[idx];
}

// ---------------------------------------------------------------------------
// Conv layer as WMMA GEMM: wave = 16 pixels (M) x ALL out-channels (COUT/16
// accumulator tiles). The A fragment is loaded once per K-chunk and reused
// across all N-tiles (back-to-back WMMAs, 4x less L2 A-traffic for COUT=64).
// Weights staged in LDS (pre-swizzled). 3x3 SAME padding via lane select;
// EXEC is all-ones at every WMMA.
// ---------------------------------------------------------------------------
template <int TAPS, int CIN, int COUT, bool RELU>
__global__ void __launch_bounds__(256) conv_wmma_kernel(
    const _Float16* __restrict__ in, const _Float16* __restrict__ wswz,
    const float* __restrict__ bias, _Float16* __restrict__ out) {
  constexpr int K      = TAPS * CIN;
  constexpr int NCHUNK = K / 32;
  constexpr int NN     = COUT / 16;
  constexpr long NM    = NPIX / 16;

  __shared__ _Float16 sW[K * COUT];
  for (int i = threadIdx.x; i < K * COUT / 8; i += 256)
    ((v8h*)sW)[i] = ((const v8h*)wswz)[i];
  __syncthreads();

  int lane = threadIdx.x & 31;
  int wave = threadIdx.x >> 5;
  long mtile = (long)blockIdx.x * 8 + wave;
  if (mtile >= NM) return;                     // wave-uniform
  long p0 = mtile * 16;
  int  w0 = (int)(p0 % W);
  int  hh = (int)((p0 / W) % H);
  int  bb = (int)(p0 / ((long)W * H));
  int  m  = lane & 15;
  int  hi = lane >> 4;
  int  wm = w0 + m;                            // W%16==0: row-uniform tile

  v8f acc[NN];
#pragma unroll
  for (int nt = 0; nt < NN; ++nt) acc[nt] = {};

#pragma unroll
  for (int ch = 0; ch < NCHUNK; ++ch) {
    const int kc  = ch * 32;
    const int tap = kc / CIN;
    const int cb  = kc % CIN;
    const int dy  = (TAPS == 9) ? (tap / 3 - 1) : 0;
    const int dx  = (TAPS == 9) ? (tap % 3 - 1) : 0;
    int hr = hh + dy, wc = wm + dx;
    bool inb = (TAPS == 1) ||
               ((unsigned)hr < (unsigned)H && (unsigned)wc < (unsigned)W);
    int hrc = min(max(hr, 0), H - 1);
    int wcc = min(max(wc, 0), W - 1);
    const _Float16* ap =
        in + (((long)bb * H + hrc) * W + wcc) * CIN + cb + hi * 8;
    v8h alo = *(const v8h*)(ap);        // K_local = hi*8 + e        (VGPR 0-3)
    v8h ahi = *(const v8h*)(ap + 16);   // K_local = 16 + hi*8 + e   (VGPR 4-7)
    v16h a;
#pragma unroll
    for (int e = 0; e < 8; ++e) { a[e] = alo[e]; a[e + 8] = ahi[e]; }
    if (!inb) {
#pragma unroll
      for (int e = 0; e < 16; ++e) a[e] = (_Float16)0;   // zero-pad tap
    }
#pragma unroll
    for (int nt = 0; nt < NN; ++nt) {
      v16h bf = *(const v16h*)&sW[(((long)ch * NN + nt) * 32 + lane) * 16];
      acc[nt] = __builtin_amdgcn_wmma_f32_16x16x32_f16(
          false, a, false, bf, (short)0, acc[nt], false, false);
    }
  }

  long pbase = p0 + hi * 8;                    // D: lanes 16-31 hold M=r+8
#pragma unroll
  for (int nt = 0; nt < NN; ++nt) {
    int n = nt * 16 + (lane & 15);
    float bn = bias[n];
#pragma unroll
    for (int r = 0; r < 8; ++r) {
      float v = acc[nt][r] + bn;
      if (RELU) v = fmaxf(v, 0.f);
      out[(pbase + r) * COUT + n] = (_Float16)v;
    }
  }
}

// ---------------------------------------------------------------------------
// Final 1x1 conv 16->2 (too skinny for WMMA) fused with flow add.
// ---------------------------------------------------------------------------
__global__ void __launch_bounds__(256) final_flow_kernel(
    const _Float16* __restrict__ act, const float* __restrict__ w4,
    const float* __restrict__ b4, const float* __restrict__ flw,
    float* __restrict__ flowc) {
  long p = (long)blockIdx.x * blockDim.x + threadIdx.x;
  if (p >= NPIX) return;
  v8h a0 = *(const v8h*)(act + p * 16);
  v8h a1 = *(const v8h*)(act + p * 16 + 8);
  float acc0 = b4[0], acc1 = b4[1];
#pragma unroll
  for (int k = 0; k < 8; ++k) {
    float x0 = (float)a0[k], x1 = (float)a1[k];
    acc0 += x0 * w4[k * 2 + 0] + x1 * w4[(k + 8) * 2 + 0];
    acc1 += x0 * w4[k * 2 + 1] + x1 * w4[(k + 8) * 2 + 1];
  }
  flowc[p * 2 + 0] = flw[p * 2 + 0] + acc0;
  flowc[p * 2 + 1] = flw[p * 2 + 1] + acc1;
}

// ---------------------------------------------------------------------------
// jax.image.resize bilinear 2x (half-pixel centers, edge clamp)
// ---------------------------------------------------------------------------
__global__ void __launch_bounds__(256) upsample_kernel(
    const float* __restrict__ flowc, float* __restrict__ out) {
  long idx = (long)blockIdx.x * blockDim.x + threadIdx.x;
  long tot = (long)BN * HO * WO;
  if (idx >= tot) return;
  int ox = (int)(idx % WO);
  int oy = (int)((idx / WO) % HO);
  int bb = (int)(idx / ((long)WO * HO));
  float sy = (oy + 0.5f) * 0.5f - 0.5f;
  float sx = (ox + 0.5f) * 0.5f - 0.5f;
  float fy = floorf(sy), fx = floorf(sx);
  float wy = sy - fy, wx = sx - fx;
  int y0 = min(max((int)fy, 0), H - 1), y1 = min(max((int)fy + 1, 0), H - 1);
  int x0 = min(max((int)fx, 0), W - 1), x1 = min(max((int)fx + 1, 0), W - 1);
  const float* base = flowc + (long)bb * H * W * 2;
#pragma unroll
  for (int c = 0; c < 2; ++c) {
    float v00 = base[((long)y0 * W + x0) * 2 + c];
    float v01 = base[((long)y0 * W + x1) * 2 + c];
    float v10 = base[((long)y1 * W + x0) * 2 + c];
    float v11 = base[((long)y1 * W + x1) * 2 + c];
    float t = v00 + wx * (v01 - v00);
    float b = v10 + wx * (v11 - v10);
    out[((long)(bb * HO + oy) * WO + ox) * 2 + c] = t + wy * (b - t);
  }
}

// ---------------------------------------------------------------------------
extern "C" void kernel_launch(void* const* d_in, const int* in_sizes, int n_in,
                              void* d_out, int out_size, void* d_ws,
                              size_t ws_size, hipStream_t stream) {
  (void)in_sizes; (void)n_in; (void)out_size; (void)ws_size;

  const float* input_1  = (const float*)d_in[0];
  const float* input_2  = (const float*)d_in[1];
  const float* flow_1_2 = (const float*)d_in[2];
  const float* flow_2_1 = (const float*)d_in[3];
  // setup_inputs dict order: w12_i @4+4i, b12_i @5+4i, w21_i @6+4i, b21_i @7+4i

  char* ws = (char*)d_ws;
  const int wK[4]    = {576, 288, 64, 64};     // TAPS*CIN per WMMA layer
  const int wCout[4] = {32, 64, 64, 16};
  _Float16* wsz[2][4];
  size_t off = 0;
  for (int d = 0; d < 2; ++d)
    for (int l = 0; l < 4; ++l) {
      wsz[d][l] = (_Float16*)(ws + off);
      size_t bytes = (size_t)wK[l] * wCout[l] * 2;
      off += (bytes + 255) & ~(size_t)255;
    }
  _Float16* actA = (_Float16*)(ws + (size_t)(1 << 20));
  _Float16* actB = (_Float16*)(ws + (size_t)(1 << 20) + 67108864ull);
  float* flowc[2] = {
      (float*)(ws + (size_t)(1 << 20) + 2ull * 67108864ull),
      (float*)(ws + (size_t)(1 << 20) + 2ull * 67108864ull + 4194304ull)};
  float* out = (float*)d_out;

  // Weight convert + swizzle (tiny)
  for (int d = 0; d < 2; ++d)
    for (int l = 0; l < 4; ++l) {
      const float* wsrc = (const float*)d_in[4 + 4 * l + 2 * d];
      int nel = wK[l] * wCout[l];
      weight_swizzle_kernel<<<(nel + 255) / 256, 256, 0, stream>>>(
          wsrc, wsz[d][l], wK[l], wCout[l]);
    }

  constexpr unsigned CONV_BLOCKS = (unsigned)((NPIX / 16) / 8);  // 4096

  for (int d = 0; d < 2; ++d) {
    const float* src = d ? input_2 : input_1;
    const float* oth = d ? input_1 : input_2;
    const float* flw = d ? flow_2_1 : flow_1_2;
    const float* b0 = (const float*)d_in[5 + 0 + 2 * d];
    const float* b1 = (const float*)d_in[5 + 4 + 2 * d];
    const float* b2 = (const float*)d_in[5 + 8 + 2 * d];
    const float* b3 = (const float*)d_in[5 + 12 + 2 * d];
    const float* w4 = (const float*)d_in[4 + 16 + 2 * d];
    const float* b4 = (const float*)d_in[5 + 16 + 2 * d];

    warp_concat_kernel<<<(unsigned)((NPIX * 4) / 256), 256, 0, stream>>>(
        src, oth, flw, actA);

    conv_wmma_kernel<9, 64, 32, true>
        <<<CONV_BLOCKS, 256, 0, stream>>>(actA, wsz[d][0], b0, actB);
    conv_wmma_kernel<9, 32, 64, true>
        <<<CONV_BLOCKS, 256, 0, stream>>>(actB, wsz[d][1], b1, actA);
    conv_wmma_kernel<1, 64, 64, true>
        <<<CONV_BLOCKS, 256, 0, stream>>>(actA, wsz[d][2], b2, actB);
    conv_wmma_kernel<1, 64, 16, true>
        <<<CONV_BLOCKS, 256, 0, stream>>>(actB, wsz[d][3], b3, actA);

    final_flow_kernel<<<(unsigned)(NPIX / 256), 256, 0, stream>>>(
        actA, w4, b4, flw, flowc[d]);
    warp_out_kernel<<<(unsigned)((NPIX * 4) / 256), 256, 0, stream>>>(
        src, flowc[d], out + (long)d * NPIX * C);
    upsample_kernel<<<(unsigned)(((long)BN * HO * WO) / 256), 256, 0, stream>>>(
        flowc[d], out + 2 * NPIX * C + (long)d * BN * HO * WO * 2);
  }
}
